// MCCDecoderAttention_22290880266513
// MI455X (gfx1250) — matrix operand order repaired and checked by reading, not compiled, not run on hardware
//
#include <hip/hip_runtime.h>
#include <hip/hip_bf16.h>

// ---------------------------------------------------------------------------
// MCC decoder attention for MI455X (gfx1250, wave32, WMMA f16 -> f32)
// B=2, N=2048, C=768, H=12, d=64, scale = 1/8
// Software-pipelined: loads for step n+1 issued before WMMAs of step n.
// ---------------------------------------------------------------------------

typedef _Float16 half_t;
typedef __attribute__((ext_vector_type(16))) _Float16 v16h;
typedef __attribute__((ext_vector_type(8)))  _Float16 v8h;
typedef __attribute__((ext_vector_type(8)))  float    v8f;

#define BB   2
#define NN   2048
#define CC   768
#define HH   12
#define DD   64
#define BH   (BB * HH)        // 24
#define MTOT (BB * NN)        // 4096
#define C3   (3 * CC)         // 2304

static __device__ __forceinline__ v16h combine8(v8h lo, v8h hi) {
    return __builtin_shufflevector(lo, hi, 0, 1, 2, 3, 4, 5, 6, 7,
                                           8, 9, 10, 11, 12, 13, 14, 15);
}

// A fragment (16x32, MxK): lane = row (l&15); K chunks at ko and ko+16, ko = l<16?0:8
static __device__ __forceinline__ v16h load_a_frag(const half_t* base, int ld,
                                                   int row0, int k0, int lane) {
    int m  = lane & 15;
    int ko = (lane < 16) ? 0 : 8;
    const half_t* p = base + (size_t)(row0 + m) * ld + k0 + ko;
    v8h lo = *(const v8h*)(p);
    v8h hi = *(const v8h*)(p + 16);
    return combine8(lo, hi);
}

// B fragment (32x16, KxN): lane = col (l&15); K chunks at ko and ko+16, ko = l<16?0:8
static __device__ __forceinline__ v16h load_b_frag(const half_t* base, int ld,
                                                   int col0, int k0, int lane) {
    int n  = lane & 15;
    int ko = (lane < 16) ? 0 : 8;
    const half_t* p = base + (size_t)(col0 + n) * ld + k0 + ko;
    return combine8(*(const v8h*)(p), *(const v8h*)(p + 16));
}

static __device__ __forceinline__ v8f wmma_f16(v16h a, v16h b, v8f c) {
    return __builtin_amdgcn_wmma_f32_16x16x32_f16(false, a, false, b,
                                                  (short)0, c, false, false);
}

// ---------------------------------------------------------------------------
// Kernel 0: f32 -> f16 conversion
// ---------------------------------------------------------------------------
__global__ void mcc_to_f16(const float* __restrict__ src,
                           half_t* __restrict__ dst, int n) {
    int i = blockIdx.x * blockDim.x + threadIdx.x;
    if (i < n) dst[i] = (half_t)src[i];
}

// ---------------------------------------------------------------------------
// Kernel 1: QKV GEMM  (M=4096, K=768, N=2304), 32x64 tile per wave,
// double-buffered fragments; scatter into Q/K/Vt layouts:
//   q,k : [BH][N][64]   vt : [BH][64][N]
// ---------------------------------------------------------------------------
__global__ __launch_bounds__(128) void mcc_qkv_gemm(
    const half_t* __restrict__ x,      // [4096][768]
    const half_t* __restrict__ w,      // [2304][768]
    half_t* __restrict__ qws, half_t* __restrict__ kws,
    half_t* __restrict__ vtws) {
    int lane = threadIdx.x & 31;
    int wave = threadIdx.x >> 5;
    int row0 = blockIdx.x * 32;
    int col0 = (blockIdx.y * 4 + wave) * 64;

    v8f acc[8] = {};   // [t*2 + rowHalf]

    v16h aL = load_a_frag(x, CC, row0,      0, lane);
    v16h aH = load_a_frag(x, CC, row0 + 16, 0, lane);
    v16h b0 = load_b_frag(w, CC, col0,      0, lane);
    v16h b1 = load_b_frag(w, CC, col0 + 16, 0, lane);
    v16h b2 = load_b_frag(w, CC, col0 + 32, 0, lane);
    v16h b3 = load_b_frag(w, CC, col0 + 48, 0, lane);

    for (int k0 = 0; k0 < CC; k0 += 32) {
        int kn = k0 + 32;
        v16h aLn, aHn, b0n, b1n, b2n, b3n;
        if (kn < CC) {                       // issue next-step loads first
            aLn = load_a_frag(x, CC, row0,      kn, lane);
            aHn = load_a_frag(x, CC, row0 + 16, kn, lane);
            b0n = load_b_frag(w, CC, col0,      kn, lane);
            b1n = load_b_frag(w, CC, col0 + 16, kn, lane);
            b2n = load_b_frag(w, CC, col0 + 32, kn, lane);
            b3n = load_b_frag(w, CC, col0 + 48, kn, lane);
        }
        acc[0] = wmma_f16(aL, b0, acc[0]);
        acc[1] = wmma_f16(aH, b0, acc[1]);
        acc[2] = wmma_f16(aL, b1, acc[2]);
        acc[3] = wmma_f16(aH, b1, acc[3]);
        acc[4] = wmma_f16(aL, b2, acc[4]);
        acc[5] = wmma_f16(aH, b2, acc[5]);
        acc[6] = wmma_f16(aL, b3, acc[6]);
        acc[7] = wmma_f16(aH, b3, acc[7]);
        if (kn < CC) {
            aL = aLn; aH = aHn;
            b0 = b0n; b1 = b1n; b2 = b2n; b3 = b3n;
        }
    }

    int nloc = lane & 15;
    int hi8  = (lane < 16) ? 0 : 8;
#pragma unroll
    for (int t = 0; t < 4; ++t) {
        int c   = col0 + t * 16 + nloc;
        int s   = c / CC;
        int rem = c % CC;
        int h   = rem >> 6;
        int dd  = rem & 63;
#pragma unroll
        for (int rh = 0; rh < 2; ++rh) {
#pragma unroll
            for (int r = 0; r < 8; ++r) {
                int m = row0 + rh * 16 + r + hi8;
                int b = m >> 11;            // /N
                int n = m & (NN - 1);
                int bh = b * HH + h;
                half_t val = (half_t)acc[t * 2 + rh][r];
                if (s == 0)
                    qws[((size_t)bh * NN + n) * DD + dd] = val;
                else if (s == 1)
                    kws[((size_t)bh * NN + n) * DD + dd] = val;
                else
                    vtws[((size_t)bh * DD + dd) * NN + n] = val;
            }
        }
    }
}

// ---------------------------------------------------------------------------
// Kernel 2: flash attention. One wave = one 16-row Q tile of one (b,h).
// Streams 32 keys/iter: 4 QK wmma + online softmax + 4 PV wmma.
// V-frag loads and next-chunk K-frag loads are issued before the softmax
// VALU block so the matrix pipes never wait on memory.
// ---------------------------------------------------------------------------
__global__ __launch_bounds__(128) void mcc_flash_attn(
    const half_t* __restrict__ q, const half_t* __restrict__ k,
    const half_t* __restrict__ vt, const int* __restrict__ unseen_p,
    half_t* __restrict__ aout) {
    __shared__ half_t lds_p[4][16 * 32];

    int lane = threadIdx.x & 31;
    int wave = threadIdx.x >> 5;
    int gw   = blockIdx.x * 4 + wave;      // 0..3071
    int bh   = gw >> 7;                    // / (N/16)
    int qb   = (gw & 127) * 16;

    const half_t* qp = q  + (size_t)bh * NN * DD;
    const half_t* kp = k  + (size_t)bh * NN * DD;
    const half_t* vp = vt + (size_t)bh * DD * NN;
    half_t* pl = lds_p[wave];

    int first_masked = NN - *unseen_p;

    int nloc = lane & 15;
    int hi8  = (lane < 16) ? 0 : 8;

    v16h qa0 = load_a_frag(qp, DD, qb, 0, lane);
    v16h qa1 = load_a_frag(qp, DD, qb, 32, lane);

    // preload K fragments for first chunk
    v16h kf00 = load_b_frag(kp, DD, 0,  0,  lane);
    v16h kf10 = load_b_frag(kp, DD, 16, 0,  lane);
    v16h kf01 = load_b_frag(kp, DD, 0,  32, lane);
    v16h kf11 = load_b_frag(kp, DD, 16, 32, lane);

    v8f  o[4] = {};
    float M[8], L[8];
#pragma unroll
    for (int r = 0; r < 8; ++r) { M[r] = -1e30f; L[r] = 0.0f; }

    for (int kb = 0; kb < NN; kb += 32) {
        // ---- S = (Q Kt), two 16-key subtiles ----
        v8f s0 = {}, s1 = {};
        s0 = wmma_f16(qa0, kf00, s0);
        s1 = wmma_f16(qa0, kf10, s1);
        s0 = wmma_f16(qa1, kf01, s0);
        s1 = wmma_f16(qa1, kf11, s1);

        // ---- issue V frags (this chunk) + K frags (next chunk) now;
        //      softmax VALU below hides their latency ----
        v16h vf0 = load_b_frag(vp, NN, 0,  kb, lane);
        v16h vf1 = load_b_frag(vp, NN, 16, kb, lane);
        v16h vf2 = load_b_frag(vp, NN, 32, kb, lane);
        v16h vf3 = load_b_frag(vp, NN, 48, kb, lane);
        int kbn = kb + 32;
        if (kbn < NN) {
            __builtin_prefetch(kp + (size_t)(kbn + 32) * DD, 0, 3);
            kf00 = load_b_frag(kp, DD, kbn,      0,  lane);
            kf10 = load_b_frag(kp, DD, kbn + 16, 0,  lane);
            kf01 = load_b_frag(kp, DD, kbn,      32, lane);
            kf11 = load_b_frag(kp, DD, kbn + 16, 32, lane);
        }

        // ---- mask + online softmax (row reductions across 16 lanes) ----
        int key0 = kb + nloc;
        int key1 = key0 + 16;
        float p0v[8], p1v[8], alpha[8];
#pragma unroll
        for (int r = 0; r < 8; ++r) {
            int qrow = qb + r + hi8;
            float a0 = s0[r] * 0.125f;
            float a1 = s1[r] * 0.125f;
            if (key0 >= first_masked && key0 != qrow) a0 = -1e30f;
            if (key1 >= first_masked && key1 != qrow) a1 = -1e30f;
            float mr = fmaxf(a0, a1);
#pragma unroll
            for (int off = 1; off < 16; off <<= 1)
                mr = fmaxf(mr, __shfl_xor(mr, off, 16));
            float mnew = fmaxf(M[r], mr);
            float al   = __expf(M[r] - mnew);
            M[r]       = mnew;
            alpha[r]   = al;
            float p0 = __expf(a0 - mnew);
            float p1 = __expf(a1 - mnew);
            float rs = p0 + p1;
#pragma unroll
            for (int off = 1; off < 16; off <<= 1)
                rs += __shfl_xor(rs, off, 16);
            L[r] = L[r] * al + rs;
            p0v[r] = p0; p1v[r] = p1;
        }

        // ---- transpose P (C-layout -> A-layout) through LDS ----
#pragma unroll
        for (int r = 0; r < 8; ++r) {
            pl[(r + hi8) * 32 + nloc]      = (half_t)p0v[r];
            pl[(r + hi8) * 32 + 16 + nloc] = (half_t)p1v[r];
        }
        asm volatile("s_wait_dscnt 0x0" ::: "memory");
        v16h pa;
        {
            int ko = (lane < 16) ? 0 : 8;
            const half_t* pp = pl + nloc * 32 + ko;
            v8h lo = *(const v8h*)(pp);
            v8h hi = *(const v8h*)(pp + 16);
            pa = combine8(lo, hi);
        }

        // ---- O = diag(alpha) * O + P @ V ----
#pragma unroll
        for (int t = 0; t < 4; ++t) {
#pragma unroll
            for (int r = 0; r < 8; ++r) o[t][r] *= alpha[r];
        }
        o[0] = wmma_f16(pa, vf0, o[0]);
        o[1] = wmma_f16(pa, vf1, o[1]);
        o[2] = wmma_f16(pa, vf2, o[2]);
        o[3] = wmma_f16(pa, vf3, o[3]);
    }

    // ---- normalize + store into [B][N][C] (f16) ----
    int b = bh / HH, h = bh % HH;
#pragma unroll
    for (int r = 0; r < 8; ++r) {
        float inv = 1.0f / L[r];
        int qrow  = qb + r + hi8;
        size_t rowbase = ((size_t)b * NN + qrow) * CC + h * DD;
#pragma unroll
        for (int t = 0; t < 4; ++t)
            aout[rowbase + t * 16 + nloc] = (half_t)(o[t][r] * inv);
    }
}

// ---------------------------------------------------------------------------
// Kernel 3: output projection  out = A @ Wp^T + bias   (M=4096, K=768, N=768)
// 32x64 tile per wave, double-buffered fragments.
// ---------------------------------------------------------------------------
__global__ __launch_bounds__(128) void mcc_proj_gemm(
    const half_t* __restrict__ a_in,   // [4096][768]
    const half_t* __restrict__ w,      // [768][768]
    const float* __restrict__ bias,    // [768]
    float* __restrict__ out) {
    int lane = threadIdx.x & 31;
    int wave = threadIdx.x >> 5;
    int row0 = blockIdx.x * 32;
    int col0 = (blockIdx.y * 4 + wave) * 64;

    v8f acc[8] = {};

    v16h aL = load_a_frag(a_in, CC, row0,      0, lane);
    v16h aH = load_a_frag(a_in, CC, row0 + 16, 0, lane);
    v16h b0 = load_b_frag(w, CC, col0,      0, lane);
    v16h b1 = load_b_frag(w, CC, col0 + 16, 0, lane);
    v16h b2 = load_b_frag(w, CC, col0 + 32, 0, lane);
    v16h b3 = load_b_frag(w, CC, col0 + 48, 0, lane);

    for (int k0 = 0; k0 < CC; k0 += 32) {
        int kn = k0 + 32;
        v16h aLn, aHn, b0n, b1n, b2n, b3n;
        if (kn < CC) {
            aLn = load_a_frag(a_in, CC, row0,      kn, lane);
            aHn = load_a_frag(a_in, CC, row0 + 16, kn, lane);
            b0n = load_b_frag(w, CC, col0,      kn, lane);
            b1n = load_b_frag(w, CC, col0 + 16, kn, lane);
            b2n = load_b_frag(w, CC, col0 + 32, kn, lane);
            b3n = load_b_frag(w, CC, col0 + 48, kn, lane);
        }
        acc[0] = wmma_f16(aL, b0, acc[0]);
        acc[1] = wmma_f16(aH, b0, acc[1]);
        acc[2] = wmma_f16(aL, b1, acc[2]);
        acc[3] = wmma_f16(aH, b1, acc[3]);
        acc[4] = wmma_f16(aL, b2, acc[4]);
        acc[5] = wmma_f16(aH, b2, acc[5]);
        acc[6] = wmma_f16(aL, b3, acc[6]);
        acc[7] = wmma_f16(aH, b3, acc[7]);
        if (kn < CC) {
            aL = aLn; aH = aHn;
            b0 = b0n; b1 = b1n; b2 = b2n; b3 = b3n;
        }
    }

    int nloc = lane & 15;
    int hi8  = (lane < 16) ? 0 : 8;
#pragma unroll
    for (int t = 0; t < 4; ++t) {
        int c   = col0 + t * 16 + nloc;
        float bv = bias[c];
#pragma unroll
        for (int rh = 0; rh < 2; ++rh) {
#pragma unroll
            for (int r = 0; r < 8; ++r) {
                int m = row0 + rh * 16 + r + hi8;
                out[(size_t)m * CC + c] = acc[t * 2 + rh][r] + bv;
            }
        }
    }
}

// ---------------------------------------------------------------------------
// Host launch
// ---------------------------------------------------------------------------
extern "C" void kernel_launch(void* const* d_in, const int* in_sizes, int n_in,
                              void* d_out, int out_size, void* d_ws, size_t ws_size,
                              hipStream_t stream) {
    const float* x      = (const float*)d_in[0];   // [2][2048][768]
    const float* w_qkv  = (const float*)d_in[1];   // [2304][768]
    const float* w_proj = (const float*)d_in[2];   // [768][768]
    const float* b_proj = (const float*)d_in[3];   // [768]
    const int*   unseen = (const int*)d_in[4];     // scalar
    float*       out    = (float*)d_out;

    char* ws = (char*)d_ws;
    size_t off = 0;
    auto carve = [&](size_t bytes) -> char* {
        char* p = ws + off;
        off = (off + bytes + 255) & ~(size_t)255;
        return p;
    };

    const size_t n_x  = (size_t)MTOT * CC;   // 3,145,728
    const size_t n_wq = (size_t)C3 * CC;     // 1,769,472
    const size_t n_wp = (size_t)CC * CC;     //   589,824
    const size_t n_hd = (size_t)BH * NN * DD;

    half_t* xh   = (half_t*)carve(n_x  * sizeof(half_t));
    half_t* wqh  = (half_t*)carve(n_wq * sizeof(half_t));
    half_t* wph  = (half_t*)carve(n_wp * sizeof(half_t));
    half_t* qws  = (half_t*)carve(n_hd * sizeof(half_t));
    half_t* kws  = (half_t*)carve(n_hd * sizeof(half_t));
    half_t* vtws = (half_t*)carve(n_hd * sizeof(half_t));
    half_t* aout = (half_t*)carve(n_x  * sizeof(half_t));

    mcc_to_f16<<<(int)((n_x  + 255) / 256), 256, 0, stream>>>(x,      xh,  (int)n_x);
    mcc_to_f16<<<(int)((n_wq + 255) / 256), 256, 0, stream>>>(w_qkv,  wqh, (int)n_wq);
    mcc_to_f16<<<(int)((n_wp + 255) / 256), 256, 0, stream>>>(w_proj, wph, (int)n_wp);

    // QKV: 128 row tiles (32 rows) x (2304 cols / (4 waves * 64 cols))
    mcc_qkv_gemm<<<dim3(MTOT / 32, C3 / 256), 128, 0, stream>>>(xh, wqh, qws, kws, vtws);

    // Attention: 24 (b,h) * 128 Q tiles = 3072 waves / 4 per block
    mcc_flash_attn<<<(BH * (NN / 16)) / 4, 128, 0, stream>>>(qws, kws, vtws, unseen, aout);

    // Projection: 128 row tiles x (768 cols / 256)
    mcc_proj_gemm<<<dim3(MTOT / 32, CC / 256), 128, 0, stream>>>(aout, wph, b_proj, out);
}
